// BahdanauAttention_89266600280590
// MI455X (gfx1250) — matrix operand order repaired and checked by reading, not compiled, not run on hardware
//
#include <hip/hip_runtime.h>
#include <hip/hip_bf16.h>

// Problem constants
#define Bn 64
#define Tn 4096
#define Hn 256
#define Un 256

typedef __attribute__((ext_vector_type(16))) __bf16 v16bf;
typedef __attribute__((ext_vector_type(8)))  __bf16 bf8v;
typedef __attribute__((ext_vector_type(8)))  float  v8f;

// f32 staging tile: 16 rows x 256 cols + 4-float row pad -> 1040B row stride
// (bank step of 4 per row -> conflict-free ds_load_b128 fragment reads)
#define FPAD 260           // floats per row
#define TILE_BYTES (16 * FPAD * 4)   // 16640 B per buffer

// ---------------------------------------------------------------------------
// Kernel 1: q_proj[b,u] = sum_h query[b,h] * W1[h,u] + b1[u]
// ---------------------------------------------------------------------------
__global__ __launch_bounds__(256) void k_qproj(const float* __restrict__ query,
                                               const float* __restrict__ W1,
                                               const float* __restrict__ b1,
                                               float* __restrict__ qproj) {
    const int b = blockIdx.x;
    const int u = threadIdx.x;
    float acc = b1[u];
    const float* qrow = query + b * Hn;
#pragma unroll 8
    for (int h = 0; h < Hn; ++h) {
        acc += qrow[h] * W1[h * Un + u];
    }
    qproj[b * Un + u] = acc;
}

// ---------------------------------------------------------------------------
// Kernel 2: convert W2 (f32 row-major [H][U]) to bf16 row-major.
// B-fragment layout: lane holds u-column (u0+lane%16), 16 k-rows (stride Un).
// ---------------------------------------------------------------------------
__global__ __launch_bounds__(256) void k_packW2(const float* __restrict__ W2,
                                                __bf16* __restrict__ w2bf) {
    const int idx = blockIdx.x * 256 + threadIdx.x;
    w2bf[idx] = (__bf16)W2[idx];
}

// ---------------------------------------------------------------------------
// Kernel 3 (main): per block -> 64 t-rows (4 tiles of 16) of one batch.
// Async DMA (global -> LDS, ASYNCcnt) double-buffers f32 value tiles while
// the 8 waves run bf16 WMMA on the previous tile.
// scores[b,t] = tanh(qproj[b,:] + values[b,t,:]@W2 + b2) . V + bV
// ---------------------------------------------------------------------------
__global__ __launch_bounds__(256) void k_scores(const float* __restrict__ values,
                                                const __bf16* __restrict__ w2bf,
                                                const float* __restrict__ qproj,
                                                const float* __restrict__ b2,
                                                const float* __restrict__ Vvec,
                                                const float* __restrict__ bV,
                                                float* __restrict__ attnOut) {
    __shared__ __align__(16) float stage[2][16 * FPAD];  // double-buffered f32 tiles
    __shared__ float sPart[8][16];                       // per-wave row partial sums

    const int tid = threadIdx.x;
    const int b   = blockIdx.y;
    const int bt0 = blockIdx.x * 64;     // first t-row of this block

    const int lane = tid & 31;
    const int wave = tid >> 5;
    const int half = lane >> 4;          // 0: rows 0-7 halves, 1: rows 8-15 halves
    const int mrow = lane & 15;
    const int u0a  = wave * 32;
    const int u0b  = u0a + 16;

    const unsigned ldsBase = (unsigned)(unsigned long long)&stage[0][0];

    // issue 16 rows x 1KB of values[b, t0:t0+16, :] into staging buffer bufIdx
    auto issue_tile = [&](int tt, int bufIdx) {
        const char* gbase =
            (const char*)(values + ((size_t)b * Tn + bt0 + tt * 16) * Hn);
        const unsigned lbase = ldsBase + (unsigned)bufIdx * TILE_BYTES;
#pragma unroll
        for (int it = 0; it < 4; ++it) {
            const int e   = tid + it * 256;     // 1024 b128 transfers per tile
            const int row = e >> 6;
            const int c16 = (e & 63) * 16;      // byte column
            const unsigned loff = lbase + (unsigned)(row * (FPAD * 4) + c16);
            const char* g = gbase + row * (Hn * 4) + c16;
            asm volatile("global_load_async_to_lds_b128 %0, %1, off"
                         :: "v"(loff), "v"(g) : "memory");
        }
    };

    issue_tile(0, 0);

    const float* qpB = qproj + b * Un;

    for (int tt = 0; tt < 4; ++tt) {
        if (tt < 3) {
            issue_tile(tt + 1, (tt + 1) & 1);
            // in-order completion: <=4 outstanding means tile tt has landed
            asm volatile("s_wait_asynccnt 0x4" ::: "memory");
        } else {
            asm volatile("s_wait_asynccnt 0x0" ::: "memory");
        }
        __syncthreads();   // tile tt visible to all waves

        const float* arow = &stage[tt & 1][0] + mrow * FPAD + half * 8;

        v8f acc0 = {};
        v8f acc1 = {};
#pragma unroll
        for (int ks = 0; ks < 8; ++ks) {
            const int k0 = ks * 32;

            // A fragment (16x32 bf16) built from f32 staging tile:
            // lane(0-15)=row M, K = koff..koff+7 and 16+koff..16+koff+7
            const float4* p = (const float4*)(arow + k0);
            float4 f0 = p[0];
            float4 f1 = p[1];
            const float4* q = (const float4*)(arow + k0 + 16);
            float4 f2 = q[0];
            float4 f3 = q[1];
            v16bf af;
            af[0]  = (__bf16)f0.x; af[1]  = (__bf16)f0.y;
            af[2]  = (__bf16)f0.z; af[3]  = (__bf16)f0.w;
            af[4]  = (__bf16)f1.x; af[5]  = (__bf16)f1.y;
            af[6]  = (__bf16)f1.z; af[7]  = (__bf16)f1.w;
            af[8]  = (__bf16)f2.x; af[9]  = (__bf16)f2.y;
            af[10] = (__bf16)f2.z; af[11] = (__bf16)f2.w;
            af[12] = (__bf16)f3.x; af[13] = (__bf16)f3.y;
            af[14] = (__bf16)f3.z; af[15] = (__bf16)f3.w;

            // B fragments (32x16 bf16): lane(0-15)=u col N, K = koff..koff+7 and 16+koff..16+koff+7
            const __bf16* w2lo = w2bf + (size_t)(k0 + half * 8) * Un + mrow;
            const __bf16* w2hi = w2bf + (size_t)(k0 + 16 + half * 8) * Un + mrow;
            v16bf bf0, bf1;
#pragma unroll
            for (int j = 0; j < 8; ++j) {
                bf0[j] = w2lo[j * Un + u0a]; bf0[j + 8] = w2hi[j * Un + u0a];
                bf1[j] = w2lo[j * Un + u0b]; bf1[j + 8] = w2hi[j * Un + u0b];
            }

            acc0 = __builtin_amdgcn_wmma_f32_16x16x32_bf16(false, af, false, bf0,
                                                           (short)0, acc0, false, false);
            acc1 = __builtin_amdgcn_wmma_f32_16x16x32_bf16(false, af, false, bf1,
                                                           (short)0, acc1, false, false);
        }

        // ---- epilogue: tanh(proj + qb) * V[u], reduce over u ----
        float rowsum[8];
#pragma unroll
        for (int r = 0; r < 8; ++r) rowsum[r] = 0.0f;

#pragma unroll
        for (int ut = 0; ut < 2; ++ut) {
            const int u    = (ut ? u0b : u0a) + mrow;
            const float qb = qpB[u] + b2[u];
            const float vv = Vvec[u];
            v8f acc = ut ? acc1 : acc0;
#pragma unroll
            for (int r = 0; r < 8; ++r) {
                float x = tanhf(acc[r] + qb);
                float p2 = x * vv;
                p2 += __shfl_xor(p2, 8);
                p2 += __shfl_xor(p2, 4);
                p2 += __shfl_xor(p2, 2);
                p2 += __shfl_xor(p2, 1);
                rowsum[r] += p2;
            }
        }

        if (mrow == 0) {
            const int base = half * 8;  // lane0 -> rows 0-7, lane16 -> rows 8-15
#pragma unroll
            for (int r = 0; r < 8; ++r) sPart[wave][base + r] = rowsum[r];
        }
        __syncthreads();

        if (tid < 16) {
            float s = 0.0f;
#pragma unroll
            for (int w = 0; w < 8; ++w) s += sPart[w][tid];
            attnOut[(size_t)b * Tn + bt0 + tt * 16 + tid] = s + bV[0];
        }
        __syncthreads();   // protect sPart + staging buffer reuse
    }
}

// ---------------------------------------------------------------------------
// Kernel 4: softmax over T per batch, in place.
// ---------------------------------------------------------------------------
__global__ __launch_bounds__(256) void k_softmax(float* __restrict__ attn) {
    __shared__ float red[256];
    const int b   = blockIdx.x;
    const int tid = threadIdx.x;
    float* row = attn + (size_t)b * Tn;

    float vals[16];
    float m = -3.402823e38f;
#pragma unroll
    for (int i = 0; i < 16; ++i) {
        vals[i] = row[tid + i * 256];
        m = fmaxf(m, vals[i]);
    }
    red[tid] = m;
    __syncthreads();
    for (int s = 128; s > 0; s >>= 1) {
        if (tid < s) red[tid] = fmaxf(red[tid], red[tid + s]);
        __syncthreads();
    }
    m = red[0];
    __syncthreads();

    float sum = 0.0f;
#pragma unroll
    for (int i = 0; i < 16; ++i) {
        vals[i] = __expf(vals[i] - m);
        sum += vals[i];
    }
    red[tid] = sum;
    __syncthreads();
    for (int s = 128; s > 0; s >>= 1) {
        if (tid < s) red[tid] += red[tid + s];
        __syncthreads();
    }
    const float inv = 1.0f / red[0];
#pragma unroll
    for (int i = 0; i < 16; ++i) row[tid + i * 256] = vals[i] * inv;
}

// ---------------------------------------------------------------------------
// Kernel 5: context partials: ctxp[b][chunk][h] = sum_{t in chunk} attn*values
// ---------------------------------------------------------------------------
__global__ __launch_bounds__(256) void k_ctx_partial(const float* __restrict__ values,
                                                     const float* __restrict__ attn,
                                                     float* __restrict__ ctxp) {
    const int chunk = blockIdx.x;   // 0..15, 256 t each
    const int b     = blockIdx.y;
    const int h     = threadIdx.x;
    const int tbase = chunk * 256;
    const float* vbase = values + ((size_t)b * Tn + tbase) * Hn + h;
    const float* abase = attn + (size_t)b * Tn + tbase;
    float acc = 0.0f;
#pragma unroll 4
    for (int i = 0; i < 256; ++i) {
        acc += abase[i] * vbase[(size_t)i * Hn];
    }
    ctxp[((size_t)b * 16 + chunk) * Hn + h] = acc;
}

// ---------------------------------------------------------------------------
// Kernel 6: fixed-order reduction of partials -> context[b][h]
// ---------------------------------------------------------------------------
__global__ __launch_bounds__(256) void k_ctx_reduce(const float* __restrict__ ctxp,
                                                    float* __restrict__ ctx) {
    const int b = blockIdx.x;
    const int h = threadIdx.x;
    float acc = 0.0f;
#pragma unroll
    for (int c = 0; c < 16; ++c) acc += ctxp[((size_t)b * 16 + c) * Hn + h];
    ctx[b * Hn + h] = acc;
}

// ---------------------------------------------------------------------------
// Launcher
// ---------------------------------------------------------------------------
extern "C" void kernel_launch(void* const* d_in, const int* in_sizes, int n_in,
                              void* d_out, int out_size, void* d_ws, size_t ws_size,
                              hipStream_t stream) {
    const float* query  = (const float*)d_in[0];   // [B,H]
    const float* values = (const float*)d_in[1];   // [B,T,H]
    const float* W1     = (const float*)d_in[2];   // [H,U]
    const float* b1     = (const float*)d_in[3];   // [U]
    const float* W2     = (const float*)d_in[4];   // [H,U]
    const float* b2     = (const float*)d_in[5];   // [U]
    const float* Vvec   = (const float*)d_in[6];   // [U,1]
    const float* bV     = (const float*)d_in[7];   // [1]

    float* outCtx  = (float*)d_out;                // [B,H]
    float* outAttn = outCtx + Bn * Hn;             // [B,T,1] flat

    // workspace layout
    char* ws = (char*)d_ws;
    float*  qproj = (float*)ws;                          // 64*256*4   = 64 KB
    __bf16* w2bf  = (__bf16*)(ws + 65536);               // 256*256*2  = 128 KB
    float*  ctxp  = (float*)(ws + 65536 + 131072);       // 64*16*256*4 = 1 MB

    k_qproj<<<Bn, 256, 0, stream>>>(query, W1, b1, qproj);
    k_packW2<<<(Hn * Un) / 256, 256, 0, stream>>>(W2, w2bf);

    dim3 gScores(Tn / 64, Bn);   // (64, 64)
    k_scores<<<gScores, 256, 0, stream>>>(values, w2bf, qproj, b2, Vvec, bV, outAttn);

    k_softmax<<<Bn, 256, 0, stream>>>(outAttn);

    dim3 gCtx(16, Bn);
    k_ctx_partial<<<gCtx, 256, 0, stream>>>(values, outAttn, ctxp);
    k_ctx_reduce<<<Bn, 256, 0, stream>>>(ctxp, outCtx);
}
